// KAN_13589276524651
// MI455X (gfx1250) — matrix operand order repaired and checked by reading, not compiled, not run on hardware
//
#include <hip/hip_runtime.h>
#include <math.h>

typedef __attribute__((ext_vector_type(16))) _Float16 v16h;
typedef __attribute__((ext_vector_type(8)))  _Float16 v8h;
typedef __attribute__((ext_vector_type(2)))  _Float16 v2h;
typedef __attribute__((ext_vector_type(8)))  float    v8f;

#define NPTS   32768
#define MTILE  64          // rows per block
#define HS     256         // hbuf row stride (floats)
#define ES     304         // ebuf row stride (halves): 288 data + 16 pad
#define O_HID  256

#define SMEM_HBUF_BYTES (MTILE * HS * 4)      // 65536
#define SMEM_EBUF_BYTES (MTILE * ES * 2)      // 38912
#define SMEM_RED_BYTES  (MTILE * 4 * 4)       // 1024
#define SMEM_TOTAL (SMEM_HBUF_BYTES + 2 * SMEM_EBUF_BYTES + SMEM_RED_BYTES) // 144384

// ---------------------------------------------------------------------------
// Cubic B-spline bases on the uniform knot grid g[i] = 0.4*i - 2.2 (12 knots).
// Matches reference: order-0 indicator then 3 Cox-de-Boor rounds (8 outputs).
// ---------------------------------------------------------------------------
__device__ __forceinline__ void bspline8(float v, float b[8]) {
    float b0[11], b1[10], b2[9];
#pragma unroll
    for (int i = 0; i < 11; ++i) {
        float gl = 0.4f * i - 2.2f;
        float gr = 0.4f * (i + 1) - 2.2f;
        b0[i] = (v >= gl && v < gr) ? 1.0f : 0.0f;
    }
#pragma unroll
    for (int i = 0; i < 10; ++i) {
        float gl = 0.4f * i - 2.2f;
        float g2 = 0.4f * (i + 2) - 2.2f;
        b1[i] = (v - gl) * 2.5f * b0[i] + (g2 - v) * 2.5f * b0[i + 1];
    }
#pragma unroll
    for (int i = 0; i < 9; ++i) {
        float gl = 0.4f * i - 2.2f;
        float g3 = 0.4f * (i + 3) - 2.2f;
        b2[i] = (v - gl) * 1.25f * b1[i] + (g3 - v) * 1.25f * b1[i + 1];
    }
#pragma unroll
    for (int i = 0; i < 8; ++i) {
        float gl = 0.4f * i - 2.2f;
        float g4 = 0.4f * (i + 4) - 2.2f;
        b[i] = (v - gl) * (1.0f / 1.2f) * b2[i] + (g4 - v) * (1.0f / 1.2f) * b2[i + 1];
    }
}

__device__ __forceinline__ float silu_f(float v) {
    return v / (1.0f + __expf(-v));
}

// ---------------------------------------------------------------------------
// Expand one 32-feature chunk of hbuf into the f16 E tile (LDS), j-major K:
//   klocal = j*32 + f_local,  j=0 -> silu, j=1..8 -> basis j-1.
// 64 rows x 16 feature-pairs = 1024 items, 256 threads x 4.
// ---------------------------------------------------------------------------
__device__ __forceinline__ void expand_chunk(const float* __restrict__ hbuf,
                                             _Float16* __restrict__ E,
                                             int cf, int tid) {
#pragma unroll
    for (int p = 0; p < 4; ++p) {
        int it = tid + p * 256;
        int m  = it >> 4;
        int fp = (it & 15) << 1;                 // even feature index
        const float* hr = hbuf + m * HS + cf * 32;
        float v0 = hr[fp], v1 = hr[fp + 1];
        float bb0[8], bb1[8];
        bspline8(v0, bb0);
        bspline8(v1, bb1);
        _Float16* ep = E + m * ES;
        v2h pr;
        pr[0] = (_Float16)silu_f(v0);
        pr[1] = (_Float16)silu_f(v1);
        *(v2h*)(ep + fp) = pr;                   // j = 0 plane
#pragma unroll
        for (int j = 1; j <= 8; ++j) {
            v2h pb;
            pb[0] = (_Float16)bb0[j - 1];
            pb[1] = (_Float16)bb1[j - 1];
            *(v2h*)(ep + j * 32 + fp) = pb;      // 4B-aligned paired store
        }
    }
}

// ---------------------------------------------------------------------------
// One hidden KAN layer (O = 256) via V_WMMA_F32_16X16X32_F16.
// Wave w: rows [16*(w>>1), +16), cols [128*(w&1), +128) -> 8 accum tiles.
// A fragment (16x32 f16): lane l (m=l&15, hi=l>>4) holds K = 16hi..16hi+15
// (contiguous, matching B). B fragment: lane holds K=16hi+j,
// pre-swizzled contiguously in the prepped weight buffer.
// ---------------------------------------------------------------------------
template <int F>
__device__ __forceinline__ void layer_wmma(const _Float16* __restrict__ wsw,
                                           float* hbuf, _Float16* eb0,
                                           _Float16* eb1, int tid) {
    constexpr int C = F / 32;
    const int wave = tid >> 5, lane = tid & 31;
    const int rg = wave >> 1, cg = wave & 1;
    const int n15 = lane & 15, hi = lane >> 4;

    v8f acc[8];
#pragma unroll
    for (int t = 0; t < 8; ++t)
#pragma unroll
        for (int r = 0; r < 8; ++r) acc[t][r] = 0.0f;

    expand_chunk(hbuf, eb0, 0, tid);

    for (int c = 0; c < C; ++c) {
        __syncthreads();
        if (c + 1 < C)                                     // pipeline: expand next
            expand_chunk(hbuf, ((c + 1) & 1) ? eb1 : eb0, c + 1, tid);
        const _Float16* E = (c & 1) ? eb1 : eb0;
        const _Float16* arow = E + (16 * rg + n15) * ES;
#pragma unroll
        for (int ks = 0; ks < 9; ++ks) {
            v8h alo = *(const v8h*)(arow + ks * 32 + 16 * hi);
            v8h ahi = *(const v8h*)(arow + ks * 32 + 16 * hi + 8);
            v16h a = __builtin_shufflevector(alo, ahi, 0, 1, 2, 3, 4, 5, 6, 7,
                                             8, 9, 10, 11, 12, 13, 14, 15);
            const _Float16* bbase =
                wsw + (size_t)(c * 9 + ks) * O_HID * 32 + (size_t)hi * 16;
#pragma unroll
            for (int ct = 0; ct < 8; ++ct) {
                const _Float16* bp = bbase + (size_t)(cg * 128 + ct * 16 + n15) * 32;
                v8h blo = *(const v8h*)bp;
                v8h bhi = *(const v8h*)(bp + 8);
                v16h b = __builtin_shufflevector(blo, bhi, 0, 1, 2, 3, 4, 5, 6,
                                                 7, 8, 9, 10, 11, 12, 13, 14, 15);
                acc[ct] = __builtin_amdgcn_wmma_f32_16x16x32_f16(
                    false, a, false, b, (short)0, acc[ct], false, false);
            }
        }
    }

    __syncthreads();
    // C/D layout: lane holds rows M = r + 8*hi, col N = lane&15.
#pragma unroll
    for (int ct = 0; ct < 8; ++ct) {
        int col = cg * 128 + ct * 16 + n15;
#pragma unroll
        for (int r = 0; r < 8; ++r)
            hbuf[(16 * rg + 8 * hi + r) * HS + col] = acc[ct][r];
    }
    __syncthreads();
}

// ---------------------------------------------------------------------------
// Final 256 -> 1 layer: VALU dot (WMMA would waste 15/16 of the tile).
// 4 threads per row, 64 features each, LDS tree reduce.
// ---------------------------------------------------------------------------
__device__ __forceinline__ void layer_out(const float* hbuf,
                                          const float* __restrict__ bw,
                                          const float* __restrict__ sw,
                                          const float* __restrict__ ss,
                                          float* red, float* __restrict__ out,
                                          int row0, int tid) {
    int m = tid >> 2, q = tid & 3;
    float a = 0.0f;
    const float* hr = hbuf + m * HS;
    for (int f = q * 64; f < q * 64 + 64; ++f) {
        float v = hr[f];
        float b[8];
        bspline8(v, b);
        float t = silu_f(v) * bw[f];
        float sc = ss[f];
#pragma unroll
        for (int cidx = 0; cidx < 8; ++cidx) t += b[cidx] * (sw[f * 8 + cidx] * sc);
        a += t;
    }
    red[tid] = a;
    __syncthreads();
    if (q == 0)
        out[row0 + m] = red[tid] + red[tid + 1] + red[tid + 2] + red[tid + 3];
}

// ---------------------------------------------------------------------------
// Fused forward: encoding + 3 WMMA layers + head. Activations stay in LDS.
// ---------------------------------------------------------------------------
extern "C" __global__ void __launch_bounds__(256)
kan_fused(const float* __restrict__ x, const float* __restrict__ freq,
          const _Float16* __restrict__ w1, const _Float16* __restrict__ w2,
          const _Float16* __restrict__ w3, const float* __restrict__ bw3,
          const float* __restrict__ sw3, const float* __restrict__ ss3,
          float* __restrict__ out) {
    extern __shared__ char smem[];
    float*    hbuf = (float*)smem;
    _Float16* eb0  = (_Float16*)(smem + SMEM_HBUF_BYTES);
    _Float16* eb1  = eb0 + MTILE * ES;
    float*    red  = (float*)(smem + SMEM_HBUF_BYTES + 2 * SMEM_EBUF_BYTES);

    const int tid  = threadIdx.x;
    const int row0 = blockIdx.x * MTILE;

    // positional encoding: h0[m][f] = f<16 ? sin(x*freq[f]) : cos(x*freq[f-16])
#pragma unroll
    for (int p = 0; p < 8; ++p) {
        int it = tid + p * 256;
        int m = it >> 5, f = it & 31;
        float e = x[row0 + m] * freq[f & 15];
        hbuf[m * HS + f] = (f < 16) ? sinf(e) : cosf(e);
    }
    __syncthreads();

    layer_wmma<32>(w1, hbuf, eb0, eb1, tid);
    layer_wmma<256>(w2, hbuf, eb0, eb1, tid);
    layer_wmma<256>(w3, hbuf, eb0, eb1, tid);
    layer_out(hbuf, bw3, sw3, ss3, red, out, row0, tid);
}

// ---------------------------------------------------------------------------
// Weight prep: W'[o][kk] with kk = (cf*9 + j)*32 + u, f = cf*32 + u;
// j==0 -> bw[o][f], j>=1 -> sw[o][f][j-1]*ss[o][f].  Stored f16 so a lane's
// B fragment ((kt*256 + o)*32 + 16*hi .. +16) is contiguous. O is always 256.
// ---------------------------------------------------------------------------
extern "C" __global__ void kan_prep(const float* __restrict__ bw,
                                    const float* __restrict__ sw,
                                    const float* __restrict__ ss,
                                    _Float16* __restrict__ wsw, int F,
                                    int total) {
    int idx = blockIdx.x * 256 + threadIdx.x;
    if (idx >= total) return;
    int u  = idx & 31;
    int o  = (idx >> 5) & 255;
    int kt = idx >> 13;
    int j  = kt % 9;
    int cf = kt / 9;
    int f  = cf * 32 + u;
    float w = (j == 0) ? bw[o * F + f]
                       : sw[(o * F + f) * 8 + (j - 1)] * ss[o * F + f];
    wsw[idx] = (_Float16)w;
}

extern "C" void kernel_launch(void* const* d_in, const int* in_sizes, int n_in,
                              void* d_out, int out_size, void* d_ws,
                              size_t ws_size, hipStream_t stream) {
    (void)in_sizes; (void)n_in; (void)out_size; (void)ws_size;
    const float* x    = (const float*)d_in[0];
    const float* freq = (const float*)d_in[1];
    const float* bw0  = (const float*)d_in[2];
    const float* sw0  = (const float*)d_in[3];
    const float* ss0  = (const float*)d_in[4];
    const float* bw1  = (const float*)d_in[5];
    const float* sw1  = (const float*)d_in[6];
    const float* ss1  = (const float*)d_in[7];
    const float* bw2  = (const float*)d_in[8];
    const float* sw2  = (const float*)d_in[9];
    const float* ss2  = (const float*)d_in[10];
    const float* bw3  = (const float*)d_in[11];
    const float* sw3  = (const float*)d_in[12];
    const float* ss3  = (const float*)d_in[13];
    float* out = (float*)d_out;

    _Float16* w1 = (_Float16*)d_ws;              // 9*256*32   = 73728 halves
    _Float16* w2 = w1 + 9 * 256 * 32;            // 72*256*32  = 589824 halves
    _Float16* w3 = w2 + 72 * 256 * 32;

    kan_prep<<<(73728 + 255) / 256, 256, 0, stream>>>(bw0, sw0, ss0, w1, 32, 73728);
    kan_prep<<<(589824 + 255) / 256, 256, 0, stream>>>(bw1, sw1, ss1, w2, 256, 589824);
    kan_prep<<<(589824 + 255) / 256, 256, 0, stream>>>(bw2, sw2, ss2, w3, 256, 589824);

    (void)hipFuncSetAttribute((const void*)kan_fused,
                              hipFuncAttributeMaxDynamicSharedMemorySize,
                              SMEM_TOTAL);
    kan_fused<<<NPTS / MTILE, 256, SMEM_TOTAL, stream>>>(x, freq, w1, w2, w3,
                                                         bw3, sw3, ss3, out);
}
